// LSTM_47218870452444
// MI455X (gfx1250) — compile-verified, matrix operand order checked by reading, not run on hardware
//
#include <hip/hip_runtime.h>
#include <hip/hip_bf16.h>

#ifndef __has_builtin
#define __has_builtin(x) 0
#endif

#define B_SZ 4096
#define T_SZ 512
#define F_SZ 16
#define G_SZ 16          // 4*HID gate columns
#define TILES_PER_WAVE 8 // 16-row WMMA tiles per wave in pass 1

typedef float    v2f  __attribute__((ext_vector_type(2)));
typedef float    v8f  __attribute__((ext_vector_type(8)));
typedef _Float16 v16h __attribute__((ext_vector_type(16)));

__device__ __forceinline__ float fast_sigmoid(float x) {
    // safe at extremes: exp(+inf)->inf => 0 ; exp(-inf)->0 => 1
    return 1.0f / (1.0f + __expf(-x));
}
__device__ __forceinline__ float fast_tanh(float x) {
    // 2*sigmoid(2x)-1, safe at extremes
    return 2.0f / (1.0f + __expf(-2.0f * x)) - 1.0f;
}

// ---------------------------------------------------------------------------
// Pass 1: xg[B*T, 16] = x[B*T, 16] @ Wk[16, 16] + b   via f32 WMMA.
// One wave = one 16-row tile per loop iteration. x is streamed with
// non-temporal loads (read-once) so the 134MB xg output stays resident in the
// 192MB L2 for pass 2. Bias folded into the C initializer (uniform per
// column N == lane%16 across all 8 C VGPRs).
// ---------------------------------------------------------------------------
__global__ void lstm_xproj_wmma(const float* __restrict__ x,
                                const float* __restrict__ Wk,
                                const float* __restrict__ bias,
                                float* __restrict__ xg)
{
    const int lane = threadIdx.x & 31;
    const int wave = (blockIdx.x * blockDim.x + threadIdx.x) >> 5;
    const int m  = lane & 15;   // row (A) / col N (B,C)
    const int hh = lane >> 4;   // half selector
    const float bval = bias[m];
    const long rowBase = (long)wave * (TILES_PER_WAVE * 16);

#if __has_builtin(__builtin_amdgcn_wmma_f32_16x16x4_f32)
    // B matrix (Wk) per K-chunk of 4: lane (m,hh) holds (K=kk, N=m),(K=kk+1, N=m)
    v2f bm[4];
#pragma unroll
    for (int k = 0; k < 4; ++k) {
        const int kk = k * 4 + 2 * hh;
        bm[k].x = Wk[kk * G_SZ + m];
        bm[k].y = Wk[(kk + 1) * G_SZ + m];
    }
    for (int tile = 0; tile < TILES_PER_WAVE; ++tile) {
        const long r0 = rowBase + tile * 16;
        v8f c;
#pragma unroll
        for (int i = 0; i < 8; ++i) c[i] = bval;
#pragma unroll
        for (int k = 0; k < 4; ++k) {
            const int kk = k * 4 + 2 * hh;
            const float* ap = x + (r0 + m) * F_SZ + kk;
            v2f a;
            a.x = __builtin_nontemporal_load(ap);
            a.y = __builtin_nontemporal_load(ap + 1);
            c = __builtin_amdgcn_wmma_f32_16x16x4_f32(
                    false, a, false, bm[k], (short)0, c, false, false);
        }
        // C layout: VGPR r, lane(m,hh) -> element (M = r + 8*hh, N = m)
        float* op = xg + (r0 + 8 * hh) * G_SZ + m;
#pragma unroll
        for (int r = 0; r < 8; ++r) op[r * G_SZ] = c[r];
    }
#else
    // Fallback: codegen-confirmed f16 WMMA, K=32 with zero padding (K>=16).
    // B (32x16): hh=0 element s = Wk row s at N=m; hh=1 half is zero (A is 0 at s>=8: rows 8..15 drop out).
    v16h bm;
#pragma unroll
    for (int i = 0; i < 16; ++i) bm[i] = (_Float16)0.0f;
    if (hh == 0) {
#pragma unroll
        for (int k = 0; k < 16; ++k) bm[k] = (_Float16)Wk[k * G_SZ + m];
    }
    for (int tile = 0; tile < TILES_PER_WAVE; ++tile) {
        const long r0 = rowBase + tile * 16;
        v8f c;
#pragma unroll
        for (int i = 0; i < 8; ++i) c[i] = bval;
        // A (16x32, 16-bit): lane(m,hh) elements 0..7 hold K = 8*hh + 0..7
        v16h a;
#pragma unroll
        for (int i = 0; i < 16; ++i) a[i] = (_Float16)0.0f;
        const float* ap = x + (r0 + m) * F_SZ + 8 * hh;
#pragma unroll
        for (int i = 0; i < 8; ++i) a[i] = (_Float16)__builtin_nontemporal_load(ap + i);
        c = __builtin_amdgcn_wmma_f32_16x16x32_f16(
                false, a, false, bm, (short)0, c, false, false);
        float* op = xg + (r0 + 8 * hh) * G_SZ + m;
#pragma unroll
        for (int r = 0; r < 8; ++r) op[r * G_SZ] = c[r];
    }
#endif
}

// ---------------------------------------------------------------------------
// Pass 2: sequential LSTM scan + MLP head. 4 lanes cooperate per batch row
// (one lane per hidden unit); h-state broadcast via __shfl within lane quads.
// xg reads are last-use -> non-temporal (hit L2, then discard).
// ---------------------------------------------------------------------------
__global__ void lstm_scan_head(const float* __restrict__ xg,
                               const float* __restrict__ Wr,
                               const float* __restrict__ W1,
                               const float* __restrict__ b1,
                               const float* __restrict__ W2,
                               const float* __restrict__ b2,
                               float* __restrict__ out)
{
    const int tid = blockIdx.x * blockDim.x + threadIdx.x;
    const int b = tid >> 2;
    const int j = tid & 3;          // hidden unit handled by this lane
    if (b >= B_SZ) return;

    // recurrent weights feeding this lane's 4 gate columns (j, j+4, j+8, j+12)
    float wri[4], wrf[4], wrg[4], wro[4];
#pragma unroll
    for (int u = 0; u < 4; ++u) {
        wri[u] = Wr[u * G_SZ + j];
        wrf[u] = Wr[u * G_SZ + j + 4];
        wrg[u] = Wr[u * G_SZ + j + 8];
        wro[u] = Wr[u * G_SZ + j + 12];
    }

    float h0 = 0.f, h1 = 0.f, h2 = 0.f, h3 = 0.f, cst = 0.f;
    const float* xp = xg + (long)b * T_SZ * G_SZ;

    for (int t = 0; t < T_SZ; ++t) {
        __builtin_prefetch((const void*)(xp + 8 * G_SZ), 0, 1);
        float xi = __builtin_nontemporal_load(xp + j);
        float xf = __builtin_nontemporal_load(xp + j + 4);
        float xv_ = __builtin_nontemporal_load(xp + j + 8);
        float xo = __builtin_nontemporal_load(xp + j + 12);
        float gi = xi + h0 * wri[0] + h1 * wri[1] + h2 * wri[2] + h3 * wri[3];
        float gf = xf + h0 * wrf[0] + h1 * wrf[1] + h2 * wrf[2] + h3 * wrf[3];
        float gg = xv_ + h0 * wrg[0] + h1 * wrg[1] + h2 * wrg[2] + h3 * wrg[3];
        float go = xo + h0 * wro[0] + h1 * wro[1] + h2 * wro[2] + h3 * wro[3];
        float ig = fast_sigmoid(gi);
        float fg = fast_sigmoid(gf);
        float gv = fast_tanh(gg);
        float og = fast_sigmoid(go);
        cst = fg * cst + ig * gv;
        float hj = og * fast_tanh(cst);
        h0 = __shfl(hj, 0, 4);
        h1 = __shfl(hj, 1, 4);
        h2 = __shfl(hj, 2, 4);
        h3 = __shfl(hj, 3, 4);
        xp += G_SZ;
    }

    if (j == 0) {
        float z0 = b2[0], z1 = b2[1], z2 = b2[2], z3 = b2[3], z4 = b2[4];
        for (int k = 0; k < 64; ++k) {
            float a = b1[k] + h0 * W1[k] + h1 * W1[64 + k]
                            + h2 * W1[128 + k] + h3 * W1[192 + k];
            a = fmaxf(a, 0.0f);
            const float* w2 = W2 + k * 5;
            z0 += a * w2[0]; z1 += a * w2[1]; z2 += a * w2[2];
            z3 += a * w2[3]; z4 += a * w2[4];
        }
        float mx = fmaxf(fmaxf(fmaxf(z0, z1), fmaxf(z2, z3)), z4);
        float e0 = __expf(z0 - mx), e1 = __expf(z1 - mx), e2 = __expf(z2 - mx);
        float e3 = __expf(z3 - mx), e4 = __expf(z4 - mx);
        float inv = 1.0f / (e0 + e1 + e2 + e3 + e4);
        float* op = out + (long)b * 5;
        op[0] = e0 * inv; op[1] = e1 * inv; op[2] = e2 * inv;
        op[3] = e3 * inv; op[4] = e4 * inv;
    }
}

// ---------------------------------------------------------------------------
// Fallback (only if workspace too small): fully fused, one thread per batch.
// ---------------------------------------------------------------------------
__global__ void lstm_fused_fallback(const float* __restrict__ x,
                                    const float* __restrict__ Wk,
                                    const float* __restrict__ Wr,
                                    const float* __restrict__ bias,
                                    const float* __restrict__ W1,
                                    const float* __restrict__ b1,
                                    const float* __restrict__ W2,
                                    const float* __restrict__ b2,
                                    float* __restrict__ out)
{
    const int b = blockIdx.x * blockDim.x + threadIdx.x;
    if (b >= B_SZ) return;
    float h[4] = {0.f, 0.f, 0.f, 0.f};
    float c[4] = {0.f, 0.f, 0.f, 0.f};
    const float* xp = x + (long)b * T_SZ * F_SZ;
    for (int t = 0; t < T_SZ; ++t) {
        __builtin_prefetch((const void*)(xp + 8 * F_SZ), 0, 1);
        float xv[16], g[16];
#pragma unroll
        for (int f = 0; f < 16; ++f) xv[f] = __builtin_nontemporal_load(xp + f);
#pragma unroll
        for (int n = 0; n < 16; ++n) {
            float acc = bias[n];
#pragma unroll
            for (int f = 0; f < 16; ++f) acc += xv[f] * Wk[f * G_SZ + n];
#pragma unroll
            for (int u = 0; u < 4; ++u) acc += h[u] * Wr[u * G_SZ + n];
            g[n] = acc;
        }
#pragma unroll
        for (int jj = 0; jj < 4; ++jj) {
            float ig = fast_sigmoid(g[jj]);
            float fg = fast_sigmoid(g[jj + 4]);
            float gv = fast_tanh(g[jj + 8]);
            float og = fast_sigmoid(g[jj + 12]);
            c[jj] = fg * c[jj] + ig * gv;
            h[jj] = og * fast_tanh(c[jj]);
        }
        xp += F_SZ;
    }
    float z[5];
#pragma unroll
    for (int o = 0; o < 5; ++o) z[o] = b2[o];
    for (int k = 0; k < 64; ++k) {
        float a = b1[k] + h[0] * W1[k] + h[1] * W1[64 + k]
                        + h[2] * W1[128 + k] + h[3] * W1[192 + k];
        a = fmaxf(a, 0.0f);
#pragma unroll
        for (int o = 0; o < 5; ++o) z[o] += a * W2[k * 5 + o];
    }
    float mx = z[0];
#pragma unroll
    for (int o = 1; o < 5; ++o) mx = fmaxf(mx, z[o]);
    float e[5], s = 0.f;
#pragma unroll
    for (int o = 0; o < 5; ++o) { e[o] = __expf(z[o] - mx); s += e[o]; }
    float inv = 1.0f / s;
    float* op = out + (long)b * 5;
#pragma unroll
    for (int o = 0; o < 5; ++o) op[o] = e[o] * inv;
}

extern "C" void kernel_launch(void* const* d_in, const int* in_sizes, int n_in,
                              void* d_out, int out_size, void* d_ws, size_t ws_size,
                              hipStream_t stream)
{
    const float* x  = (const float*)d_in[0];
    const float* Wk = (const float*)d_in[1];
    const float* Wr = (const float*)d_in[2];
    const float* bb = (const float*)d_in[3];
    const float* W1 = (const float*)d_in[4];
    const float* b1 = (const float*)d_in[5];
    const float* W2 = (const float*)d_in[6];
    const float* b2 = (const float*)d_in[7];
    float* out = (float*)d_out;

    const size_t xg_bytes = (size_t)B_SZ * T_SZ * G_SZ * sizeof(float);
    if (ws_size >= xg_bytes) {
        float* xg = (float*)d_ws;
        // pass 1: 2,097,152 rows -> 131,072 tiles -> 16,384 waves
        const int totalRows = B_SZ * T_SZ;
        const int waves = totalRows / (16 * TILES_PER_WAVE);
        const int threads = waves * 32;
        lstm_xproj_wmma<<<threads / 256, 256, 0, stream>>>(x, Wk, bb, xg);
        // pass 2: 4 lanes per batch row = 16,384 threads, small blocks to spread
        lstm_scan_head<<<(B_SZ * 4) / 64, 64, 0, stream>>>(xg, Wr, W1, b1, W2, b2, out);
    } else {
        lstm_fused_fallback<<<B_SZ / 64, 64, 0, stream>>>(x, Wk, Wr, bb, W1, b1, W2, b2, out);
    }
}
